// LocalBandSimilarityBlock_81801947120117
// MI455X (gfx1250) — matrix operand order repaired and checked by reading, not compiled, not run on hardware
//
#include <hip/hip_runtime.h>
#include <hip/hip_bf16.h>
#include <math.h>

// ---------------------------------------------------------------------------
// MI455X (gfx1250) implementation of the LocalBandSimilarityBlock.
// All GEMMs run on v_wmma_f32_16x16x32_bf16 (fp32 accumulate).
// Attention is flash-style: 16 queries/block, key strips of 128, online
// softmax, augmented K=1024 GEMM folds the cosine-similarity term in.
// ---------------------------------------------------------------------------

#define N_TOK   6144
#define DM      512
#define DFF     2048
#define QSCALE  0.04419417382415922f   // 1/sqrt(512)

typedef __attribute__((ext_vector_type(16))) __bf16          bf16x16;
typedef __attribute__((ext_vector_type(8)))  float           f32x8;
typedef __attribute__((ext_vector_type(8)))  unsigned short  u16x8;

__device__ __forceinline__ unsigned short f2bf(float f) {
  unsigned int u = __builtin_bit_cast(unsigned int, f);
  u += 0x7FFFu + ((u >> 16) & 1u);          // round-to-nearest-even
  return (unsigned short)(u >> 16);
}
__device__ __forceinline__ float bf2f(unsigned short h) {
  unsigned int u = ((unsigned int)h) << 16;
  return __builtin_bit_cast(float, u);
}

union FragU { struct { u16x8 lo, hi; } s; bf16x16 v; };

// A-fragment (16x32 bf16): lane holds row (lane&15); 16 contiguous K values
// starting at kg*16 -> two adjacent b128 loads.  p must point at col kg*16.
__device__ __forceinline__ bf16x16 frag_a(const unsigned short* p) {
  FragU u; u.s.lo = *(const u16x8*)p; u.s.hi = *(const u16x8*)(p + 8); return u.v;
}
// B-fragment (32x16 bf16): lane holds col (lane&15); 16 contiguous K values
// starting at kg*16 -> two adjacent b128 loads.
__device__ __forceinline__ bf16x16 frag_b(const unsigned short* p) {
  FragU u; u.s.lo = *(const u16x8*)p; u.s.hi = *(const u16x8*)(p + 8); return u.v;
}
__device__ __forceinline__ f32x8 wmma_bf16(bf16x16 a, bf16x16 b, f32x8 c) {
  return __builtin_amdgcn_wmma_f32_16x16x32_bf16(false, a, false, b,
                                                 (short)0, c, false, false);
}

// ---------------------------------------------------------------------------
// Weight convert + transpose: W (fin x fout) f32  ->  Wt (fout x fin) bf16
// so WMMA B-fragments read contiguous-K rows.
// ---------------------------------------------------------------------------
__global__ void cvt_wT_kernel(const float* __restrict__ W,
                              unsigned short* __restrict__ Wt,
                              int fin, int fout) {
  int idx = blockIdx.x * 256 + threadIdx.x;
  if (idx >= fin * fout) return;
  int i = idx / fout, o = idx - i * fout;          // W[i][o]
  Wt[(size_t)o * fin + i] = f2bf(W[idx]);
}

// ---------------------------------------------------------------------------
// LayerNorm row kernel.  Writes hb = bf16(h).  If writeAug, also writes the
// normalized-row halves of the augmented attention operands:
//   aaug[:,512:1024] = bf16(beta * h/||h||)   baug[:,512:1024] = bf16(h/||h||)
// One block (256 threads) per row; each thread covers cols t and t+256.
// ---------------------------------------------------------------------------
__global__ __launch_bounds__(256)
void ln_prep_kernel(const float* __restrict__ x,
                    const float* __restrict__ g,
                    const float* __restrict__ b,
                    unsigned short* __restrict__ hb,
                    unsigned short* __restrict__ aaug,
                    unsigned short* __restrict__ baug,
                    float simbeta, int writeAug) {
  __shared__ float red[256];
  int row = blockIdx.x, t = threadIdx.x;
  const float* xr = x + (size_t)row * DM;
  float x0 = xr[t], x1 = xr[t + 256];

  red[t] = x0 + x1; __syncthreads();
  for (int o = 128; o > 0; o >>= 1) { if (t < o) red[t] += red[t + o]; __syncthreads(); }
  float mu = red[0] * (1.0f / DM);
  __syncthreads();

  float d0 = x0 - mu, d1 = x1 - mu;
  red[t] = d0 * d0 + d1 * d1; __syncthreads();
  for (int o = 128; o > 0; o >>= 1) { if (t < o) red[t] += red[t + o]; __syncthreads(); }
  float rstd = rsqrtf(red[0] * (1.0f / DM) + 1e-5f);
  __syncthreads();

  float h0 = d0 * rstd * g[t]       + b[t];
  float h1 = d1 * rstd * g[t + 256] + b[t + 256];
  hb[(size_t)row * DM + t]       = f2bf(h0);
  hb[(size_t)row * DM + t + 256] = f2bf(h1);

  if (writeAug) {
    red[t] = h0 * h0 + h1 * h1; __syncthreads();
    for (int o = 128; o > 0; o >>= 1) { if (t < o) red[t] += red[t + o]; __syncthreads(); }
    float inv = 1.0f / fmaxf(sqrtf(red[0]), 1e-8f);
    size_t base = (size_t)row * 1024 + 512;
    aaug[base + t]       = f2bf(simbeta * h0 * inv);
    aaug[base + t + 256] = f2bf(simbeta * h1 * inv);
    baug[base + t]       = f2bf(h0 * inv);
    baug[base + t + 256] = f2bf(h1 * inv);
  }
}

// ---------------------------------------------------------------------------
// Generic WMMA GEMM:  C = epilogue( A(MxK bf16) @ Bt(NcxK bf16)^T , bias )
// Block tile 64(M) x 256(N); 8 waves as 2(M) x 4(N), each wave owns a
// 32x64 sub-tile (2x4 C fragments -> 8 WMMAs per 6 b128-pair loads).
// MODE: 0 store bf16, 1 store bf16 transposed (ldc = M stride),
//       2 f32 residual-add, 3 gelu -> bf16.   val = alpha*(acc + bias)
// ---------------------------------------------------------------------------
constexpr int EPI_BF16 = 0, EPI_BF16_T = 1, EPI_F32_RESADD = 2, EPI_BF16_GELU = 3;

template<int MODE>
__global__ __launch_bounds__(256)
void gemm_bf16_kernel(const unsigned short* __restrict__ A, int lda,
                      const unsigned short* __restrict__ Bt, int ldb,
                      const float* __restrict__ bias,
                      const float* __restrict__ res,
                      unsigned short* __restrict__ outB,
                      float* __restrict__ outF,
                      int ldc, int K, float alpha) {
  int wid = threadIdx.x >> 5, lane = threadIdx.x & 31;
  int laneN = lane & 15, kg = lane >> 4;
  int Mb = blockIdx.y * 64, Nb = blockIdx.x * 256;
  int wm = (wid & 1) * 32, wn = (wid >> 1) * 64;

  f32x8 c[2][4] = {};
  const unsigned short* ar[2];
  ar[0] = A + (size_t)(Mb + wm      + laneN) * lda;
  ar[1] = A + (size_t)(Mb + wm + 16 + laneN) * lda;
  const unsigned short* br[4];
  #pragma unroll
  for (int ni = 0; ni < 4; ni++)
    br[ni] = Bt + (size_t)(Nb + wn + ni * 16 + laneN) * ldb;

  for (int k0 = 0; k0 < K; k0 += 32) {
    // batch all fragment loads so the clause issues them before the wait
    bf16x16 a0 = frag_a(ar[0] + k0 + kg * 16);
    bf16x16 a1 = frag_a(ar[1] + k0 + kg * 16);
    bf16x16 b0 = frag_b(br[0] + k0 + kg * 16);
    bf16x16 b1 = frag_b(br[1] + k0 + kg * 16);
    bf16x16 b2 = frag_b(br[2] + k0 + kg * 16);
    bf16x16 b3 = frag_b(br[3] + k0 + kg * 16);
    c[0][0] = wmma_bf16(a0, b0, c[0][0]);
    c[0][1] = wmma_bf16(a0, b1, c[0][1]);
    c[0][2] = wmma_bf16(a0, b2, c[0][2]);
    c[0][3] = wmma_bf16(a0, b3, c[0][3]);
    c[1][0] = wmma_bf16(a1, b0, c[1][0]);
    c[1][1] = wmma_bf16(a1, b1, c[1][1]);
    c[1][2] = wmma_bf16(a1, b2, c[1][2]);
    c[1][3] = wmma_bf16(a1, b3, c[1][3]);
  }

  #pragma unroll
  for (int mi = 0; mi < 2; mi++)
  #pragma unroll
  for (int ni = 0; ni < 4; ni++) {
    int col = Nb + wn + ni * 16 + laneN;
    float bv = bias ? bias[col] : 0.0f;
    #pragma unroll
    for (int vj = 0; vj < 8; vj++) {
      int row = Mb + wm + mi * 16 + vj + 8 * kg;
      float v = alpha * (c[mi][ni][vj] + bv);
      if      (MODE == EPI_BF16)       outB[(size_t)row * ldc + col] = f2bf(v);
      else if (MODE == EPI_BF16_T)     outB[(size_t)col * ldc + row] = f2bf(v);
      else if (MODE == EPI_F32_RESADD) outF[(size_t)row * ldc + col] =
                                         res[(size_t)row * ldc + col] + v;
      else {
        float ge = 0.5f * v * (1.0f + erff(v * 0.70710678118654752f));
        outB[(size_t)row * ldc + col] = f2bf(ge);
      }
    }
  }
}

// ---------------------------------------------------------------------------
// Flash-style masked attention.
//   Aaug (N x 1024 bf16) = [q*scale | hn*beta]   (queries)
//   Baug (N x 1024 bf16) = [k       | hn     ]   (keys)
//   vt   (512 x N bf16)  = V^T
// Block = 16 queries, 8 waves.  Per 128-key strip: each wave computes one
// 16x16 logits tile (K=1024 -> 32 WMMAs, loads batched 2 steps ahead),
// masked online softmax with cross-wave stats in LDS, P staged in LDS, then
// each wave accumulates its 64-wide D chunk of P@V (16 WMMAs, B fragments
// hoisted).  Next strip's key/value rows are prefetched (global_prefetch).
// Isolated rows fall back to v[i].
// ---------------------------------------------------------------------------
__global__ __launch_bounds__(256)
void attn_kernel(const unsigned short* __restrict__ Aaug,
                 const unsigned short* __restrict__ Baug,
                 const unsigned short* __restrict__ vt,
                 const int* __restrict__ grd,
                 unsigned short* __restrict__ ob, int N) {
  __shared__ __align__(16) unsigned short sA[16 * 1024];  // query block, K=1024
  __shared__ __align__(16) unsigned short sP[16 * 128];   // probs strip
  __shared__ float sMax[8][16], sSum[8][16];
  __shared__ float sM[16], sL[16], sCorr[16];
  __shared__ int   sQG[16][2];

  const float NEG_INF = -__builtin_inff();
  int q0 = blockIdx.x * 16;
  int t = threadIdx.x, wid = t >> 5, lane = t & 31;
  int laneN = lane & 15, kg = lane >> 4;
  int d0 = wid * 64;

  { // stage the 16x1024 query tile into LDS (b128 copies)
    const u16x8* src = (const u16x8*)(Aaug + (size_t)q0 * 1024);
    u16x8* dst = (u16x8*)sA;
    for (int i = t; i < 2048; i += 256) dst[i] = src[i];
  }
  if (t < 16) {
    sQG[t][0] = grd[(q0 + t) * 2];
    sQG[t][1] = grd[(q0 + t) * 2 + 1];
    sM[t] = NEG_INF; sL[t] = 0.0f;
  }
  __syncthreads();

  f32x8 o[4] = {};

  for (int key0 = 0; key0 < N; key0 += 128) {
    int myk = key0 + wid * 16 + laneN;
    int kgx = grd[myk * 2], kgy = grd[myk * 2 + 1];

    const unsigned short* arow = sA + laneN * 1024;
    const unsigned short* brow = Baug + (size_t)myk * 1024;

    // prefetch next strip's key row and V rows (L2 is the home for K/V)
    if (key0 + 128 < N) {
      const unsigned short* nb = brow + (size_t)128 * 1024;
      __builtin_prefetch(nb, 0, 1);
      __builtin_prefetch(nb + 512, 0, 1);
      __builtin_prefetch(vt + (size_t)(d0 + laneN) * N + key0 + 128, 0, 1);
    }

    // ---- S = Aaug_block @ Baug_strip^T  (K = 1024) ----
    f32x8 s = {};
    #pragma unroll 4
    for (int kk = 0; kk < 1024; kk += 64) {
      bf16x16 a0 = frag_a(arow + kk      + kg * 16);
      bf16x16 b0 = frag_b(brow + kk      + kg * 16);
      bf16x16 a1 = frag_a(arow + kk + 32 + kg * 16);
      bf16x16 b1 = frag_b(brow + kk + 32 + kg * 16);
      s = wmma_bf16(a0, b0, s);
      s = wmma_bf16(a1, b1, s);
    }

    // ---- masked row-max (16-lane shuffle reduce per C row) ----
    float rmax[8];
    #pragma unroll
    for (int vj = 0; vj < 8; vj++) {
      int m = vj + 8 * kg;
      int dx = sQG[m][0] - kgx; dx = dx < 0 ? -dx : dx;
      int dy = sQG[m][1] - kgy; dy = dy < 0 ? -dy : dy;
      bool ok = (dx <= 2) && (dy <= 2) && ((q0 + m) != myk);
      float sv = ok ? s[vj] : NEG_INF;
      for (int x = 1; x < 16; x <<= 1) sv = fmaxf(sv, __shfl_xor(sv, x, 32));
      rmax[vj] = sv;
    }
    if (laneN == 0)
      for (int vj = 0; vj < 8; vj++) sMax[wid][vj + 8 * kg] = rmax[vj];
    __syncthreads();

    if (t < 16) {
      float sm = sMax[0][t];
      for (int w = 1; w < 8; w++) sm = fmaxf(sm, sMax[w][t]);
      float om = sM[t], nm = fmaxf(om, sm), cr;
      if (nm == NEG_INF)      cr = 1.0f;
      else if (om == NEG_INF) cr = 0.0f;
      else                    cr = __expf(om - nm);
      sCorr[t] = cr; sM[t] = nm;
    }
    __syncthreads();

    // ---- rescale O, compute P = exp(S - m), stage P, partial row sums ----
    #pragma unroll
    for (int vj = 0; vj < 8; vj++) {
      float cr = sCorr[vj + 8 * kg];
      o[0][vj] *= cr; o[1][vj] *= cr; o[2][vj] *= cr; o[3][vj] *= cr;
    }
    float rsum[8];
    #pragma unroll
    for (int vj = 0; vj < 8; vj++) {
      int m = vj + 8 * kg;
      int dx = sQG[m][0] - kgx; dx = dx < 0 ? -dx : dx;
      int dy = sQG[m][1] - kgy; dy = dy < 0 ? -dy : dy;
      bool ok = (dx <= 2) && (dy <= 2) && ((q0 + m) != myk);
      float nm = sM[m];
      float pv = (ok && nm > NEG_INF) ? __expf(s[vj] - nm) : 0.0f;
      sP[m * 128 + wid * 16 + laneN] = f2bf(pv);
      float sv = pv;
      for (int x = 1; x < 16; x <<= 1) sv += __shfl_xor(sv, x, 32);
      rsum[vj] = sv;
    }
    if (laneN == 0)
      for (int vj = 0; vj < 8; vj++) sSum[wid][vj + 8 * kg] = rsum[vj];
    __syncthreads();

    if (t < 16) {
      float ss = 0.0f;
      for (int w = 0; w < 8; w++) ss += sSum[w][t];
      sL[t] = sL[t] * sCorr[t] + ss;
    }

    // ---- O(16 x 64 chunk) += P(16x128) @ V(128 x 64 chunk) ----
    {
      const unsigned short* vbase =
          vt + (size_t)(d0 + laneN) * N + key0 + kg * 16;
      const size_t nistep = (size_t)16 * N;   // 16 vt rows per n-tile
      #pragma unroll
      for (int kk = 0; kk < 4; kk++) {
        bf16x16 a  = frag_a(sP + laneN * 128 + kk * 32 + kg * 16);
        const unsigned short* bp = vbase + kk * 32;
        bf16x16 b0 = frag_b(bp);
        bf16x16 b1 = frag_b(bp + nistep);
        bf16x16 b2 = frag_b(bp + 2 * nistep);
        bf16x16 b3 = frag_b(bp + 3 * nistep);
        o[0] = wmma_bf16(a, b0, o[0]);
        o[1] = wmma_bf16(a, b1, o[1]);
        o[2] = wmma_bf16(a, b2, o[2]);
        o[3] = wmma_bf16(a, b3, o[3]);
      }
    }
    __syncthreads();
  }

  // ---- epilogue: normalize; isolated rows -> v[i] ----
  #pragma unroll
  for (int ni = 0; ni < 4; ni++) {
    int dg = d0 + ni * 16 + laneN;
    #pragma unroll
    for (int vj = 0; vj < 8; vj++) {
      int m = vj + 8 * kg, qi = q0 + m;
      float l = sL[m];
      float val = (l > 0.0f) ? (o[ni][vj] / l)
                             : bf2f(vt[(size_t)dg * N + qi]);
      ob[(size_t)qi * DM + dg] = f2bf(val);
    }
  }
}

// ---------------------------------------------------------------------------
// Host launcher
// ---------------------------------------------------------------------------
extern "C" void kernel_launch(void* const* d_in, const int* in_sizes, int n_in,
                              void* d_out, int out_size, void* d_ws, size_t ws_size,
                              hipStream_t stream) {
  const float* x    = (const float*)d_in[0];
  const int*   grd  = (const int*)  d_in[1];
  const float* Wq   = (const float*)d_in[2];  const float* bq = (const float*)d_in[3];
  const float* Wk   = (const float*)d_in[4];  const float* bk = (const float*)d_in[5];
  const float* Wv   = (const float*)d_in[6];  const float* bv = (const float*)d_in[7];
  const float* Wo   = (const float*)d_in[8];  const float* bo = (const float*)d_in[9];
  const float* ln1g = (const float*)d_in[10]; const float* ln1b = (const float*)d_in[11];
  const float* ln2g = (const float*)d_in[12]; const float* ln2b = (const float*)d_in[13];
  const float* W1   = (const float*)d_in[14]; const float* b1 = (const float*)d_in[15];
  const float* W2   = (const float*)d_in[16]; const float* b2 = (const float*)d_in[17];
  float* out = (float*)d_out;

  char* w = (char*)d_ws; size_t off = 0;
  auto nextu = [&](size_t e) { unsigned short* p = (unsigned short*)(w + off);
                               off += ((e * 2) + 255) & ~(size_t)255; return p; };
  auto nextf = [&](size_t e) { float* p = (float*)(w + off);
                               off += ((e * 4) + 255) & ~(size_t)255; return p; };

  unsigned short* WqT  = nextu((size_t)DM * DM);
  unsigned short* WkT  = nextu((size_t)DM * DM);
  unsigned short* WvT  = nextu((size_t)DM * DM);
  unsigned short* WoT  = nextu((size_t)DM * DM);
  unsigned short* W1T  = nextu((size_t)DM * DFF);
  unsigned short* W2T  = nextu((size_t)DFF * DM);
  unsigned short* hb   = nextu((size_t)N_TOK * DM);
  unsigned short* Aaug = nextu((size_t)N_TOK * 1024);
  unsigned short* Baug = nextu((size_t)N_TOK * 1024);
  unsigned short* vt   = nextu((size_t)DM * N_TOK);
  unsigned short* ob   = nextu((size_t)N_TOK * DM);
  float*          x1   = nextf((size_t)N_TOK * DM);
  unsigned short* g1b  = Aaug;   // alias: Aaug/Baug dead after attention

  // weights -> bf16 transposed
  cvt_wT_kernel<<<(DM*DM + 255)/256, 256, 0, stream>>>(Wq, WqT, DM, DM);
  cvt_wT_kernel<<<(DM*DM + 255)/256, 256, 0, stream>>>(Wk, WkT, DM, DM);
  cvt_wT_kernel<<<(DM*DM + 255)/256, 256, 0, stream>>>(Wv, WvT, DM, DM);
  cvt_wT_kernel<<<(DM*DM + 255)/256, 256, 0, stream>>>(Wo, WoT, DM, DM);
  cvt_wT_kernel<<<(DM*DFF + 255)/256, 256, 0, stream>>>(W1, W1T, DM, DFF);
  cvt_wT_kernel<<<(DFF*DM + 255)/256, 256, 0, stream>>>(W2, W2T, DFF, DM);

  // LN1 + normalized-row halves of the augmented operands
  ln_prep_kernel<<<N_TOK, 256, 0, stream>>>(x, ln1g, ln1b, hb, Aaug, Baug,
                                            1.0f /*SIM_BETA*/, 1);

  dim3 gP(DM / 256, N_TOK / 64);
  // q*scale -> Aaug[:, :512]      (alpha folds the 1/sqrt(D) into q and bq)
  gemm_bf16_kernel<EPI_BF16><<<gP, 256, 0, stream>>>(
      hb, DM, WqT, DM, bq, nullptr, Aaug, nullptr, 1024, DM, QSCALE);
  // k -> Baug[:, :512]
  gemm_bf16_kernel<EPI_BF16><<<gP, 256, 0, stream>>>(
      hb, DM, WkT, DM, bk, nullptr, Baug, nullptr, 1024, DM, 1.0f);
  // v -> vt (transposed, D x N) so P@V B-fragments are contiguous
  gemm_bf16_kernel<EPI_BF16_T><<<gP, 256, 0, stream>>>(
      hb, DM, WvT, DM, bv, nullptr, vt, nullptr, N_TOK, DM, 1.0f);

  // masked flash attention
  attn_kernel<<<N_TOK / 16, 256, 0, stream>>>(Aaug, Baug, vt, grd, ob, N_TOK);

  // x1 = x + out @ Wo + bo
  gemm_bf16_kernel<EPI_F32_RESADD><<<gP, 256, 0, stream>>>(
      ob, DM, WoT, DM, bo, x, nullptr, x1, DM, DM, 1.0f);

  // LN2 (reuse hb)
  ln_prep_kernel<<<N_TOK, 256, 0, stream>>>(x1, ln2g, ln2b, hb,
                                            nullptr, nullptr, 0.0f, 0);

  // g1 = gelu(h2 @ W1 + b1)
  dim3 gF1(DFF / 256, N_TOK / 64);
  gemm_bf16_kernel<EPI_BF16_GELU><<<gF1, 256, 0, stream>>>(
      hb, DM, W1T, DM, b1, nullptr, g1b, nullptr, DFF, DM, 1.0f);

  // out = x1 + g1 @ W2 + b2
  gemm_bf16_kernel<EPI_F32_RESADD><<<gP, 256, 0, stream>>>(
      g1b, DFF, W2T, DFF, b2, x1, nullptr, out, DM, DFF, 1.0f);
}
